// MoeTokenLevelFeedForward_64596308132334
// MI455X (gfx1250) — matrix operand order repaired and checked by reading, not compiled, not run on hardware
//
#include <hip/hip_runtime.h>
#include <math.h>

#define NTOK 4096
#define DM   1024
#define DFF  2048
#define NEXP 8
#define CAP  4128            // per-expert bucket capacity (4096 + one 32-token tile)
#define TM   32              // tokens per tile
#define NTILES (CAP / TM)    // 129
#define KSTEP 32
#define BCHUNK 256           // output columns processed per pass (64 per wave)
#define ASTR 40              // LDS K-stride (elements) for A tile, 80B rows (16B aligned)
#define BSTR 40              // LDS K-stride (elements) for transposed B tile
#define HSTR (DFF + 8)       // LDS row stride for h (4112B rows, 16B aligned)
#define ZERO_FILL 2.2204460492503131e-16f

typedef __bf16 bf16;
typedef __attribute__((ext_vector_type(16))) __bf16 v16bf;
typedef __attribute__((ext_vector_type(8)))  __bf16 v8bf;
typedef __attribute__((ext_vector_type(8)))  float  v8f;

// ---------------- WMMA fragment loads (CDNA5 wave32 layouts) ----------------
// A (16x32 bf16, MxK): lane L<16 holds row M=L, K in {0..7, 16..23};
// lane L>=16 holds row M=L-16, K in {8..15, 24..31}. Two 16B LDS loads.
__device__ __forceinline__ v16bf load_a_frag(const bf16* base, int stride, int lane) {
  const bf16* p = base + (lane & 15) * stride + ((lane >> 4) << 3);
  union { v16bf v; v8bf h[2]; } u;
  u.h[0] = *(const v8bf*)p;
  u.h[1] = *(const v8bf*)(p + 16);
  return u.v;
}
// B (32x16 bf16, KxN): lane L<16 holds col N=L, K in {0..7, 16..23}; lane L>=16 holds
// col N=L-16, K in {8..15, 24..31}.  With B stored K-transposed in LDS ([n][k]) this is
// two 16B loads per lane.
__device__ __forceinline__ v16bf load_b_frag(const bf16* baseT, int stride, int lane) {
  const bf16* p = baseT + (lane & 15) * stride + ((lane >> 4) << 3);
  union { v16bf v; v8bf h[2]; } u;
  u.h[0] = *(const v8bf*)p;
  u.h[1] = *(const v8bf*)(p + 16);
  return u.v;
}

__device__ __forceinline__ uint32_t pack2_bf16(float lo, float hi) {
  union { bf16 h[2]; uint32_t u; } p;
  p.h[0] = (bf16)lo; p.h[1] = (bf16)hi;
  return p.u;   // lowers to v_cvt_pk_bf16_f32
}

__device__ __forceinline__ float softplusf(float v) {
  return v > 20.f ? v : log1pf(expf(v));
}
__device__ __forceinline__ float phi_cdf(float z) {
  return 0.5f * erfcf(-z * 0.70710678118654752f);
}

// ---------------- kernel 1: zero init ----------------
__global__ void zero_kernel(float* __restrict__ y, float* imp, float* loadv, int* counts) {
  size_t i = (size_t)blockIdx.x * blockDim.x + threadIdx.x;
  if (i < (size_t)NTOK * DM) y[i] = 0.f;
  if (blockIdx.x == 0 && threadIdx.x < NEXP) {
    imp[threadIdx.x] = 0.f;
    loadv[threadIdx.x] = 0.f;
    counts[threadIdx.x] = 0;
  }
}

// ---------------- kernel 2: noisy top-k gating + bucket dispatch ----------------
__global__ __launch_bounds__(128) void gating_kernel(
    const float* __restrict__ x, const float* __restrict__ noise,
    const float* __restrict__ wg, const float* __restrict__ wn,
    int* __restrict__ counts, int* __restrict__ btok, float* __restrict__ bgate,
    float* __restrict__ imp, float* __restrict__ loadv)
{
  const int wave = threadIdx.x >> 5;
  const int lane = threadIdx.x & 31;
  const int t = blockIdx.x * 4 + wave;

  float cg[NEXP], cn[NEXP];
#pragma unroll
  for (int j = 0; j < NEXP; ++j) { cg[j] = 0.f; cn[j] = 0.f; }

  const float* xr = x + (size_t)t * DM;
  for (int d = lane; d < DM; d += 32) {
    float xv = xr[d];
    const float* wgr = wg + d * NEXP;
    const float* wnr = wn + d * NEXP;
#pragma unroll
    for (int j = 0; j < NEXP; ++j) { cg[j] += xv * wgr[j]; cn[j] += xv * wnr[j]; }
  }
#pragma unroll
  for (int j = 0; j < NEXP; ++j) {
#pragma unroll
    for (int s = 16; s > 0; s >>= 1) {
      cg[j] += __shfl_xor(cg[j], s, 32);
      cn[j] += __shfl_xor(cn[j], s, 32);
    }
  }

  if (lane == 0) {
    float clean[NEXP], stdv[NEXP], noisy[NEXP];
#pragma unroll
    for (int j = 0; j < NEXP; ++j) {
      clean[j] = cg[j];
      stdv[j]  = softplusf(cn[j]) + 0.01f;
      noisy[j] = clean[j] + noise[(size_t)t * NEXP + j] * stdv[j];
    }
    // top-3 of noisy logits (track indices of top-2)
    float v0 = -1e30f, v1 = -1e30f, v2 = -1e30f;
    int i0 = 0, i1 = 0;
#pragma unroll
    for (int j = 0; j < NEXP; ++j) {
      float v = noisy[j];
      if (v > v0)      { v2 = v1; v1 = v0; i1 = i0; v0 = v; i0 = j; }
      else if (v > v1) { v2 = v1; v1 = v; i1 = j; }
      else if (v > v2) { v2 = v; }
    }
    // softmax over top-2
    float e1 = expf(v1 - v0);
    float g0 = 1.f / (1.f + e1);
    float g1 = e1 / (1.f + e1);
    atomicAdd(imp + i0, g0);
    atomicAdd(imp + i1, g1);
    // load via _prob_in_top_k: thr_in = 3rd noisy value, thr_out = 2nd
    float thr_in = v2, thr_out = v1;
#pragma unroll
    for (int j = 0; j < NEXP; ++j) {
      bool is_in = noisy[j] > thr_in;
      float z = (clean[j] - (is_in ? thr_in : thr_out)) / stdv[j];
      atomicAdd(loadv + j, phi_cdf(z));
    }
    // bucket insertion (order nondeterministic, result invariant)
    int s0 = atomicAdd(counts + i0, 1);
    btok[i0 * CAP + s0] = t; bgate[i0 * CAP + s0] = g0;
    int s1 = atomicAdd(counts + i1, 1);
    btok[i1 * CAP + s1] = t; bgate[i1 * CAP + s1] = g1;
  }
}

// ---------------- kernel 3: pad buckets to 32-token tiles + aux loss ----------------
__global__ void pad_loss_kernel(const float* __restrict__ imp, const float* __restrict__ loadv,
                                const int* __restrict__ counts, int* __restrict__ padded,
                                int* __restrict__ btok, float* __restrict__ bgate,
                                float* __restrict__ loss_out)
{
  int e = threadIdx.x;
  if (e < NEXP) {
    int c = counts[e];
    int p = (c + TM - 1) & ~(TM - 1);
    padded[e] = p;
    for (int s = c; s < p; ++s) { btok[e * CAP + s] = 0; bgate[e * CAP + s] = 0.f; }
  }
  if (threadIdx.x == 0) {
    float mi = 0.f, ml = 0.f;
#pragma unroll
    for (int j = 0; j < NEXP; ++j) { mi += imp[j]; ml += loadv[j]; }
    mi *= (1.f / NEXP); ml *= (1.f / NEXP);
    float vi = 0.f, vl = 0.f;
#pragma unroll
    for (int j = 0; j < NEXP; ++j) {
      float di = imp[j] - mi, dl = loadv[j] - ml;
      vi += di * di; vl += dl * dl;
    }
    vi *= (1.f / (NEXP - 1)); vl *= (1.f / (NEXP - 1));
    *loss_out = (vi / (mi * mi + 1e-10f) + vl / (ml * ml + 1e-10f)) * 0.1f;
  }
}

// ---------------- kernel 4: fused expert FFN (two chained bf16 WMMA GEMMs) ----------------
__global__ __launch_bounds__(128) void moe_ffn_kernel(
    const float* __restrict__ x,
    const float* __restrict__ w1, const float* __restrict__ b1,
    const float* __restrict__ w2, const float* __restrict__ b2,
    const int* __restrict__ padded, const int* __restrict__ btok,
    const float* __restrict__ bgate, float* __restrict__ y)
{
  __shared__ __align__(16) bf16 sA[TM * ASTR];          //  2.5 KB  gathered x tile
  __shared__ __align__(16) bf16 sB[BCHUNK * BSTR];      // 20.0 KB  K-transposed weight tile
  __shared__ __align__(16) bf16 sH[TM * HSTR];          //128.5 KB  full h tile (32 x 2048)
  __shared__ int   sTok[TM];
  __shared__ float sGate[TM];

  const int e    = blockIdx.x / NTILES;
  const int tile = blockIdx.x % NTILES;
  if (tile * TM >= padded[e]) return;       // uniform early exit for empty tiles

  const int tid  = threadIdx.x;
  const int wave = tid >> 5;
  const int lane = tid & 31;
  const int colt = lane & 15;
  const int rb   = (lane >> 4) * 8;

  if (tid < TM) {
    sTok[tid]  = btok[e * CAP + tile * TM + tid];
    sGate[tid] = bgate[e * CAP + tile * TM + tid];
  }

  // ---- phase 1: H = relu(Xg @ W1[e] + b1[e]), kept in LDS as bf16 ----
  for (int fc = 0; fc < DFF; fc += BCHUNK) {
    v8f acc[2][4];
#pragma unroll
    for (int mt = 0; mt < 2; ++mt)
#pragma unroll
      for (int nt = 0; nt < 4; ++nt)
        acc[mt][nt] = (v8f){0.f,0.f,0.f,0.f,0.f,0.f,0.f,0.f};

    for (int kk = 0; kk < DM; kk += KSTEP) {
      __syncthreads();
      // --- fill sA: 32 rows x 32 k, one float4 global load -> one b64 LDS store ---
#pragma unroll
      for (int r = 0; r < (TM * KSTEP / 4) / 128; ++r) {   // 2 trips
        int i  = tid + r * 128;
        int m  = i >> 3;
        int k4 = (i & 7) << 2;
        const float4 v = *(const float4*)&x[(size_t)sTok[m] * DM + kk + k4];
        union { uint32_t u[2]; uint64_t l; } d;
        d.u[0] = pack2_bf16(v.x, v.y);
        d.u[1] = pack2_bf16(v.z, v.w);
        *(uint64_t*)&sA[m * ASTR + k4] = d.l;
      }
      // --- fill sB (K-transposed): two adjacent K rows per trip, packed b32 stores ---
#pragma unroll
      for (int r = 0; r < ((KSTEP / 2) * (BCHUNK / 4)) / 128; ++r) {  // 8 trips
        int i  = tid + r * 128;
        int kp = i >> 6;               // k pair index 0..15 -> k = 2*kp
        int n4 = (i & 63) << 2;
        const float* g0 = &w1[((size_t)e * DM + kk + 2 * kp) * DFF + fc + n4];
        const float4 a = *(const float4*)g0;
        const float4 b = *(const float4*)(g0 + DFF);
        *(uint32_t*)&sB[(n4 + 0) * BSTR + 2 * kp] = pack2_bf16(a.x, b.x);
        *(uint32_t*)&sB[(n4 + 1) * BSTR + 2 * kp] = pack2_bf16(a.y, b.y);
        *(uint32_t*)&sB[(n4 + 2) * BSTR + 2 * kp] = pack2_bf16(a.z, b.z);
        *(uint32_t*)&sB[(n4 + 3) * BSTR + 2 * kp] = pack2_bf16(a.w, b.w);
      }
      if (kk + KSTEP < DM)  // hint next weight slice into cache
        __builtin_prefetch(&w1[((size_t)e * DM + kk + KSTEP) * DFF + fc + (tid << 3)], 0, 1);
      __syncthreads();

      v16bf a0 = load_a_frag(sA,             ASTR, lane);
      v16bf a1 = load_a_frag(sA + 16 * ASTR, ASTR, lane);
      const bf16* bb = sB + wave * 64 * BSTR;
#pragma unroll
      for (int nt = 0; nt < 4; ++nt) {
        v16bf b = load_b_frag(bb + nt * 16 * BSTR, BSTR, lane);
        acc[0][nt] = __builtin_amdgcn_wmma_f32_16x16x32_bf16(false, a0, false, b,
                                                             (short)0, acc[0][nt], false, false);
        acc[1][nt] = __builtin_amdgcn_wmma_f32_16x16x32_bf16(false, a1, false, b,
                                                             (short)0, acc[1][nt], false, false);
      }
    }
    // epilogue: bias + relu -> bf16 h in LDS
#pragma unroll
    for (int mt = 0; mt < 2; ++mt)
#pragma unroll
      for (int nt = 0; nt < 4; ++nt) {
        int col = fc + wave * 64 + nt * 16 + colt;
        float bias = b1[e * DFF + col];
#pragma unroll
        for (int i = 0; i < 8; ++i) {
          int row = mt * 16 + rb + i;
          float v = acc[mt][nt][i] + bias;
          sH[row * HSTR + col] = (bf16)(v > 0.f ? v : 0.f);
        }
      }
  }

  // ---- phase 2: y += gate * (H @ W2[e] + b2[e]) ----
  for (int nc = 0; nc < DM; nc += BCHUNK) {
    v8f acc[2][4];
#pragma unroll
    for (int mt = 0; mt < 2; ++mt)
#pragma unroll
      for (int nt = 0; nt < 4; ++nt)
        acc[mt][nt] = (v8f){0.f,0.f,0.f,0.f,0.f,0.f,0.f,0.f};

    for (int kk = 0; kk < DFF; kk += KSTEP) {
      __syncthreads();
#pragma unroll
      for (int r = 0; r < ((KSTEP / 2) * (BCHUNK / 4)) / 128; ++r) {  // 8 trips
        int i  = tid + r * 128;
        int kp = i >> 6;
        int n4 = (i & 63) << 2;
        const float* g0 = &w2[((size_t)e * DFF + kk + 2 * kp) * DM + nc + n4];
        const float4 a = *(const float4*)g0;
        const float4 b = *(const float4*)(g0 + DM);
        *(uint32_t*)&sB[(n4 + 0) * BSTR + 2 * kp] = pack2_bf16(a.x, b.x);
        *(uint32_t*)&sB[(n4 + 1) * BSTR + 2 * kp] = pack2_bf16(a.y, b.y);
        *(uint32_t*)&sB[(n4 + 2) * BSTR + 2 * kp] = pack2_bf16(a.z, b.z);
        *(uint32_t*)&sB[(n4 + 3) * BSTR + 2 * kp] = pack2_bf16(a.w, b.w);
      }
      if (kk + KSTEP < DFF)
        __builtin_prefetch(&w2[((size_t)e * DFF + kk + KSTEP) * DM + nc + (tid << 3)], 0, 1);
      __syncthreads();

      v16bf a0 = load_a_frag(sH + kk,             HSTR, lane);  // A straight from LDS h
      v16bf a1 = load_a_frag(sH + 16 * HSTR + kk, HSTR, lane);
      const bf16* bb = sB + wave * 64 * BSTR;
#pragma unroll
      for (int nt = 0; nt < 4; ++nt) {
        v16bf b = load_b_frag(bb + nt * 16 * BSTR, BSTR, lane);
        acc[0][nt] = __builtin_amdgcn_wmma_f32_16x16x32_bf16(false, a0, false, b,
                                                             (short)0, acc[0][nt], false, false);
        acc[1][nt] = __builtin_amdgcn_wmma_f32_16x16x32_bf16(false, a1, false, b,
                                                             (short)0, acc[1][nt], false, false);
      }
    }
    // epilogue: gate-scaled combine into y (token appears in 2 experts -> atomic)
#pragma unroll
    for (int mt = 0; mt < 2; ++mt)
#pragma unroll
      for (int nt = 0; nt < 4; ++nt) {
        int col = nc + wave * 64 + nt * 16 + colt;
        float bias = b2[e * DM + col];
#pragma unroll
        for (int i = 0; i < 8; ++i) {
          int row = mt * 16 + rb + i;
          float v = sGate[row] * (acc[mt][nt][i] + bias);
          atomicAdd(&y[(size_t)sTok[row] * DM + col], v);
        }
      }
  }
}

// ---------------- kernel 5: combined[combined == 0] = eps ----------------
__global__ void eps_kernel(float* __restrict__ y) {
  size_t i = (size_t)blockIdx.x * blockDim.x + threadIdx.x;
  if (i < (size_t)NTOK * DM) {
    if (y[i] == 0.f) y[i] = ZERO_FILL;
  }
}

extern "C" void kernel_launch(void* const* d_in, const int* in_sizes, int n_in,
                              void* d_out, int out_size, void* d_ws, size_t ws_size,
                              hipStream_t stream)
{
  const float* x     = (const float*)d_in[0];
  const float* noise = (const float*)d_in[1];
  const float* wg    = (const float*)d_in[2];
  const float* wn    = (const float*)d_in[3];
  const float* w1    = (const float*)d_in[4];
  const float* b1    = (const float*)d_in[5];
  const float* w2    = (const float*)d_in[6];
  const float* b2    = (const float*)d_in[7];

  float* y    = (float*)d_out;
  float* loss = y + (size_t)NTOK * DM;

  char* p = (char*)d_ws;
  int*   counts = (int*)p;   p += 256;
  int*   padded = (int*)p;   p += 256;
  float* imp    = (float*)p; p += 256;
  float* loadv  = (float*)p; p += 256;
  int*   btok   = (int*)p;   p += sizeof(int)   * NEXP * CAP;
  float* bgate  = (float*)p; /* p += sizeof(float) * NEXP * CAP; total ~265 KB */

  const size_t ytot = (size_t)NTOK * DM;
  const int zb = (int)((ytot + 255) / 256);

  zero_kernel<<<zb, 256, 0, stream>>>(y, imp, loadv, counts);
  gating_kernel<<<NTOK / 4, 128, 0, stream>>>(x, noise, wg, wn, counts, btok, bgate, imp, loadv);
  pad_loss_kernel<<<1, 32, 0, stream>>>(imp, loadv, counts, padded, btok, bgate, loss);
  moe_ffn_kernel<<<NEXP * NTILES, 128, 0, stream>>>(x, w1, b1, w2, b2, padded, btok, bgate, y);
  eps_kernel<<<zb, 256, 0, stream>>>(y);
}
